// HeteroGraphConv_61194694034258
// MI455X (gfx1250) — matrix operand order repaired and checked by reading, not compiled, not run on hardware
//
#include <hip/hip_runtime.h>
#include <hip/hip_bf16.h>
#include <math.h>

typedef __attribute__((ext_vector_type(16))) __bf16 v16bf;
typedef __attribute__((ext_vector_type(8)))  __bf16 v8bf;
typedef __attribute__((ext_vector_type(8)))  float  v8f;

#define H_DIM 256
#define HEADS 8
#define HD 32
#define LDS_STRIDE 264  // 256 + 8 bf16 pad (528B row pitch -> 4-bank rotation)

// ---------------------------------------------------------------------------
// Pre-pack a 256x256 f32 weight matrix into bf16, laid out in WMMA B-fragment
// order: element index = ((ks*16 + ct)*32 + lane)*16 + j  where
//   K = ks*32 + (lane>>4)*16 + j,  N = ct*16 + (lane&15).
// A lane's whole fragment is one contiguous 32B chunk -> 2x global_load_b128.
// ---------------------------------------------------------------------------
__global__ __launch_bounds__(256) void prepack_w(const float* __restrict__ W,
                                                 __bf16* __restrict__ Wp) {
  int i = blockIdx.x * 256 + threadIdx.x;            // 0 .. 65535
  int j    = i & 15;
  int lane = (i >> 4) & 31;
  int ctks = i >> 9;
  int ct   = ctks & 15;
  int ks   = ctks >> 4;
  int K = ks * 32 + (lane >> 4) * 16 + j;
  int N = ct * 16 + (lane & 15);
  Wp[i] = (__bf16)W[(size_t)K * H_DIM + N];
}

// ---------------------------------------------------------------------------
// GEMM: Y[nrows x 256] = X[nrows x 256] @ W[256 x 256] + bias[256]
// Block = 256 threads = 8 waves, 32-row tile staged in LDS as bf16.
// Wave w owns cols [32w,32w+32): 2 row-tiles x 2 col-tiles = 4 WMMA accums.
// Inner loop: 4x ds_load_b128 (A) + 4x global_load_b128 (B) + 4x WMMA.
// ---------------------------------------------------------------------------
__global__ __launch_bounds__(256) void gemm256_bias(
    const float* __restrict__ X, const __bf16* __restrict__ Wp,
    const float* __restrict__ bias, float* __restrict__ Y, int nrows) {
  __shared__ __bf16 As[32 * LDS_STRIDE];

  const int tid  = threadIdx.x;
  const int lane = tid & 31;
  const int wv   = tid >> 5;                 // 0..7
  const int row0 = blockIdx.x * 32;
  if (row0 >= nrows) return;

  // Cooperative stage + f32->bf16 convert of the 32x256 A tile.
  for (int i = tid; i < 32 * H_DIM; i += 256) {
    int r = i >> 8;
    int c = i & (H_DIM - 1);
    float x = (row0 + r < nrows) ? X[(size_t)(row0 + r) * H_DIM + c] : 0.0f;
    As[r * LDS_STRIDE + c] = (__bf16)x;
  }
  __syncthreads();

  const int m    = lane & 15;
  const int half = lane >> 4;
  const int ct0  = 2 * wv;                   // this wave's two column tiles
  const int ct1  = 2 * wv + 1;

  v8f c00 = {}, c01 = {}, c10 = {}, c11 = {};

#pragma unroll
  for (int ks = 0; ks < 8; ++ks) {
    const int k0 = ks * 32;
    // A fragments (contiguous k = 16*half + j, matching packed B): two 16B LDS chunks.
    v16bf a0, a1;
    {
      const __bf16* ap0 = &As[m * LDS_STRIDE + k0 + half * 16];
      const __bf16* ap1 = &As[(16 + m) * LDS_STRIDE + k0 + half * 16];
      ((v8bf*)&a0)[0] = *(const v8bf*)(ap0);
      ((v8bf*)&a0)[1] = *(const v8bf*)(ap0 + 8);
      ((v8bf*)&a1)[0] = *(const v8bf*)(ap1);
      ((v8bf*)&a1)[1] = *(const v8bf*)(ap1 + 8);
    }
    // B fragments: one contiguous 32B vector per lane from pre-packed W.
    v16bf b0 = *(const v16bf*)(Wp + (((size_t)(ks * 16 + ct0) * 32 + lane) << 4));
    v16bf b1 = *(const v16bf*)(Wp + (((size_t)(ks * 16 + ct1) * 32 + lane) << 4));

    c00 = __builtin_amdgcn_wmma_f32_16x16x32_bf16(false, a0, false, b0, (short)0, c00, false, false);
    c01 = __builtin_amdgcn_wmma_f32_16x16x32_bf16(false, a0, false, b1, (short)0, c01, false, false);
    c10 = __builtin_amdgcn_wmma_f32_16x16x32_bf16(false, a1, false, b0, (short)0, c10, false, false);
    c11 = __builtin_amdgcn_wmma_f32_16x16x32_bf16(false, a1, false, b1, (short)0, c11, false, false);
  }

  // C layout: VGPR v -> row (v + 8*half), col = lane&15.
  const int cA = ct0 * 16 + m;
  const float bias0 = bias[cA];
  const float bias1 = bias[cA + 16];
#pragma unroll
  for (int v = 0; v < 8; ++v) {
    int r0 = row0 + v + half * 8;        // row-tile 0
    int r1 = r0 + 16;                    // row-tile 1
    if (r0 < nrows) {
      Y[(size_t)r0 * H_DIM + cA]      = c00[v] + bias0;
      Y[(size_t)r0 * H_DIM + cA + 16] = c01[v] + bias1;
    }
    if (r1 < nrows) {
      Y[(size_t)r1 * H_DIM + cA]      = c10[v] + bias0;
      Y[(size_t)r1 * H_DIM + cA + 16] = c11[v] + bias1;
    }
  }
}

// ---------------------------------------------------------------------------
// Edge attention: one wave per destination node (dst sorted -> binary search,
// no atomics). Lane owns channels [8*lane, 8*lane+8), head = lane>>2.
// ---------------------------------------------------------------------------
__global__ __launch_bounds__(256) void edge_attn(
    const float* __restrict__ Q, const float* __restrict__ K,
    const float* __restrict__ V, const int* __restrict__ src,
    const int* __restrict__ dst, const float* __restrict__ ea_rel,
    float* __restrict__ attn_ws, float* __restrict__ out,
    int n_dst, int n_edges) {
  const int lane = threadIdx.x & 31;
  const int node = blockIdx.x * 8 + (threadIdx.x >> 5);
  if (node >= n_dst) return;

  int lo = 0, hi = n_edges;
  while (lo < hi) { int mid = (lo + hi) >> 1; if (dst[mid] <  node) lo = mid + 1; else hi = mid; }
  const int beg = lo;
  hi = n_edges;
  while (lo < hi) { int mid = (lo + hi) >> 1; if (dst[mid] <= node) lo = mid + 1; else hi = mid; }
  const int end = lo;

  const float scale = 0.1767766952966369f;  // 1/sqrt(32)
  float q[8];
#pragma unroll
  for (int j = 0; j < 8; ++j)
    q[j] = Q[(size_t)node * H_DIM + lane * 8 + j] * scale;
  const float ea = ea_rel[lane >> 2];

  float mmax = 0.0f;  // reference clamps segment max at 0
  for (int e = beg; e < end; ++e) {
    const float* kp = &K[(size_t)src[e] * H_DIM + lane * 8];
    float p = 0.f;
#pragma unroll
    for (int j = 0; j < 8; ++j) p += q[j] * kp[j];
    p += __shfl_xor(p, 1, 32);
    p += __shfl_xor(p, 2, 32);
    float attn = p + ea;
    if ((lane & 3) == 0) attn_ws[(size_t)e * HEADS + (lane >> 2)] = attn;
    mmax = fmaxf(mmax, attn);
  }

  float s = 0.f;
  float acc[8] = {0.f, 0.f, 0.f, 0.f, 0.f, 0.f, 0.f, 0.f};
  for (int e = beg; e < end; ++e) {
    float w = __expf(attn_ws[(size_t)e * HEADS + (lane >> 2)] - mmax);
    s += w;
    const float* vp = &V[(size_t)src[e] * H_DIM + lane * 8];
#pragma unroll
    for (int j = 0; j < 8; ++j) acc[j] += w * vp[j];
  }
  const float inv = 1.0f / fmaxf(s, 1e-9f);
#pragma unroll
  for (int j = 0; j < 8; ++j)
    out[(size_t)node * H_DIM + lane * 8 + j] = acc[j] * inv;
}

// ---------------------------------------------------------------------------
__global__ void combine2(const float* __restrict__ a, const float* __restrict__ b,
                         const float* __restrict__ w2, float* __restrict__ out,
                         size_t n) {
  size_t i = (size_t)blockIdx.x * blockDim.x + threadIdx.x;
  if (i >= n) return;
  float w0 = w2[0], w1 = w2[1];
  float mx = fmaxf(w0, w1);
  float e0 = __expf(w0 - mx), e1 = __expf(w1 - mx);
  float rs = 1.0f / (e0 + e1);
  out[i] = (e0 * rs) * a[i] + (e1 * rs) * b[i];
}

// ---------------------------------------------------------------------------
__global__ __launch_bounds__(256) void residual_ln(
    const float* __restrict__ x, const float* __restrict__ y,
    const float* __restrict__ g, const float* __restrict__ b,
    float* __restrict__ out, int nrows) {
  const int lane = threadIdx.x & 31;
  const int row  = blockIdx.x * 8 + (threadIdx.x >> 5);
  if (row >= nrows) return;

  float v[8];
  float sum = 0.f;
#pragma unroll
  for (int j = 0; j < 8; ++j) {
    size_t idx = (size_t)row * H_DIM + lane * 8 + j;
    v[j] = x[idx] + y[idx];
    sum += v[j];
  }
#pragma unroll
  for (int off = 1; off < 32; off <<= 1) sum += __shfl_xor(sum, off, 32);
  const float mu = sum * (1.0f / 256.0f);

  float var = 0.f;
#pragma unroll
  for (int j = 0; j < 8; ++j) { float d = v[j] - mu; var += d * d; }
#pragma unroll
  for (int off = 1; off < 32; off <<= 1) var += __shfl_xor(var, off, 32);
  var *= (1.0f / 256.0f);
  const float rinv = rsqrtf(var + 1e-5f);

#pragma unroll
  for (int j = 0; j < 8; ++j) {
    int c = lane * 8 + j;
    out[(size_t)row * H_DIM + c] = (v[j] - mu) * rinv * g[c] + b[c];
  }
}

// ---------------------------------------------------------------------------
extern "C" void kernel_launch(void* const* d_in, const int* in_sizes, int n_in,
                              void* d_out, int out_size, void* d_ws, size_t ws_size,
                              hipStream_t stream) {
  (void)n_in; (void)out_size; (void)ws_size;

  const float* x_paper   = (const float*)d_in[0];
  const float* x_author  = (const float*)d_in[1];
  const int*   wr_src    = (const int*)d_in[2];
  const int*   wr_dst    = (const int*)d_in[3];
  const int*   wb_src    = (const int*)d_in[4];
  const int*   wb_dst    = (const int*)d_in[5];
  const int*   ci_src    = (const int*)d_in[6];
  const int*   ci_dst    = (const int*)d_in[7];
  const float* Wq        = (const float*)d_in[8];
  const float* bq        = (const float*)d_in[9];
  const float* Wk        = (const float*)d_in[10];
  const float* bk        = (const float*)d_in[11];
  const float* Wv        = (const float*)d_in[12];
  const float* bv        = (const float*)d_in[13];
  const float* eattn     = (const float*)d_in[14];
  const float* Wout_p    = (const float*)d_in[15];
  const float* bout_p    = (const float*)d_in[16];
  const float* Wout_a    = (const float*)d_in[17];
  const float* bout_a    = (const float*)d_in[18];
  const float* ln_g_p    = (const float*)d_in[19];
  const float* ln_b_p    = (const float*)d_in[20];
  const float* ln_g_a    = (const float*)d_in[21];
  const float* ln_b_a    = (const float*)d_in[22];
  const float* w_paper   = (const float*)d_in[23];

  const int NP = in_sizes[0] / H_DIM;   // 100000
  const int NA = in_sizes[1] / H_DIM;   // 50000
  const int E  = in_sizes[2];           // 400000
  const size_t HH = (size_t)H_DIM * H_DIM;

  // Workspace layout (floats); proj buffers reused relation-by-relation.
  float* ws    = (float*)d_ws;
  float* Qbuf  = ws;                           // NP*256
  float* Kbuf  = Qbuf + (size_t)NP * H_DIM;    // NP*256
  float* Vbuf  = Kbuf + (size_t)NP * H_DIM;    // NP*256
  float* aggW  = Vbuf + (size_t)NP * H_DIM;    // NP*256  (writes -> paper)
  float* aggC  = aggW + (size_t)NP * H_DIM;    // NP*256  (cites  -> paper)
  float* aggA  = aggC + (size_t)NP * H_DIM;    // NA*256  (written-> author)
  float* attnw = aggA + (size_t)NA * H_DIM;    // E*8 per-edge logits scratch
  __bf16* WpBase = (__bf16*)(attnw + (size_t)E * HEADS);  // 11 x 64K bf16

  const dim3 blk(256);
  const int gbP = (NP + 31) / 32, gbA = (NA + 31) / 32;
  const int geP = (NP + 7) / 8,   geA = (NA + 7) / 8;

  // Slots: 0..2 Wq[r], 3..5 Wk[r], 6..8 Wv[r], 9 Wout_paper, 10 Wout_author.
  __bf16* wp[11];
  for (int i = 0; i < 11; ++i) wp[i] = WpBase + (size_t)i * HH;
  for (int r = 0; r < 3; ++r) {
    prepack_w<<<256, blk, 0, stream>>>(Wq + r * HH, wp[r]);
    prepack_w<<<256, blk, 0, stream>>>(Wk + r * HH, wp[3 + r]);
    prepack_w<<<256, blk, 0, stream>>>(Wv + r * HH, wp[6 + r]);
  }
  prepack_w<<<256, blk, 0, stream>>>(Wout_p, wp[9]);
  prepack_w<<<256, blk, 0, stream>>>(Wout_a, wp[10]);

  // --- relation 0: writes (author -> paper) ---
  gemm256_bias<<<gbP, blk, 0, stream>>>(x_paper,  wp[0], bq + 0 * H_DIM, Qbuf, NP);
  gemm256_bias<<<gbA, blk, 0, stream>>>(x_author, wp[3], bk + 0 * H_DIM, Kbuf, NA);
  gemm256_bias<<<gbA, blk, 0, stream>>>(x_author, wp[6], bv + 0 * H_DIM, Vbuf, NA);
  edge_attn<<<geP, blk, 0, stream>>>(Qbuf, Kbuf, Vbuf, wr_src, wr_dst,
                                     eattn + 0 * HEADS, attnw, aggW, NP, E);

  // --- relation 1: written_by (paper -> author) ---
  gemm256_bias<<<gbA, blk, 0, stream>>>(x_author, wp[1], bq + 1 * H_DIM, Qbuf, NA);
  gemm256_bias<<<gbP, blk, 0, stream>>>(x_paper,  wp[4], bk + 1 * H_DIM, Kbuf, NP);
  gemm256_bias<<<gbP, blk, 0, stream>>>(x_paper,  wp[7], bv + 1 * H_DIM, Vbuf, NP);
  edge_attn<<<geA, blk, 0, stream>>>(Qbuf, Kbuf, Vbuf, wb_src, wb_dst,
                                     eattn + 1 * HEADS, attnw, aggA, NA, E);

  // --- relation 2: cites (paper -> paper) ---
  gemm256_bias<<<gbP, blk, 0, stream>>>(x_paper, wp[2], bq + 2 * H_DIM, Qbuf, NP);
  gemm256_bias<<<gbP, blk, 0, stream>>>(x_paper, wp[5], bk + 2 * H_DIM, Kbuf, NP);
  gemm256_bias<<<gbP, blk, 0, stream>>>(x_paper, wp[8], bv + 2 * H_DIM, Vbuf, NP);
  edge_attn<<<geP, blk, 0, stream>>>(Qbuf, Kbuf, Vbuf, ci_src, ci_dst,
                                     eattn + 2 * HEADS, attnw, aggC, NP, E);

  // --- combine (paper: softmax over [writes, cites]) -> Qbuf (reused) ---
  {
    size_t n = (size_t)NP * H_DIM;
    combine2<<<(unsigned)((n + 255) / 256), blk, 0, stream>>>(aggW, aggC, w_paper, Qbuf, n);
  }

  // --- output projections (author combine weight = softmax of 1 elem = 1) ---
  gemm256_bias<<<gbP, blk, 0, stream>>>(Qbuf, wp[9],  bout_p, Kbuf, NP);
  gemm256_bias<<<gbA, blk, 0, stream>>>(aggA, wp[10], bout_a, Vbuf, NA);

  // --- residual + LayerNorm -> d_out (paper rows, then author rows) ---
  float* out = (float*)d_out;
  residual_ln<<<geP, blk, 0, stream>>>(x_paper,  Kbuf, ln_g_p, ln_b_p, out, NP);
  residual_ln<<<geA, blk, 0, stream>>>(x_author, Vbuf, ln_g_a, ln_b_a,
                                       out + (size_t)NP * H_DIM, NA);
}
